// S4_31387620999392
// MI455X (gfx1250) — compile-verified, matrix operand order checked by reading, not run on hardware
//
#include <hip/hip_runtime.h>
#include <hip/hip_bf16.h>

#define HID     1024
#define TSTEPS  4096
#define OUTDIM  32000
#define LAYERS  4

#define SCAN_BLOCKS  32
#define SCAN_THREADS 256

// GEMM block tile
#define BM 128
#define BN 256
#define BK 32

typedef __attribute__((ext_vector_type(16))) __bf16        v16bf;
typedef __attribute__((ext_vector_type(8)))  float         v8f;
typedef __attribute__((ext_vector_type(4)))  unsigned int  u32x4;
typedef __attribute__((ext_vector_type(8)))  int           i32x8;
typedef __attribute__((ext_vector_type(4)))  int           i32x4;

#if defined(__HIP_DEVICE_COMPILE__) && __has_builtin(__builtin_amdgcn_tensor_load_to_lds) && __has_builtin(__builtin_amdgcn_s_wait_tensorcnt)
#define TDM_OK 1
#else
#define TDM_OK 0
#endif

__device__ __forceinline__ float gelu_f(float x) {
    float x3 = x * x * x;
    return 0.5f * x * (1.0f + tanhf(0.7978845608028654f * (x + 0.044715f * x3)));
}

union FragBF { v16bf v; uint4 q[2]; };

#if TDM_OK
// ---------------------------------------------------------------------------
// Issue one TDM 2D tile load: global (row-major, row_stride_elems bf16/row)
// -> LDS at lds_off, tile = tile_d0 elems x tile_d1 rows, packed contiguously.
// D# bitfields per CDNA5 ISA (group0 / group1).
// ---------------------------------------------------------------------------
__device__ __forceinline__ void tdm_load_2d(const void* gaddr, unsigned int lds_off,
                                            unsigned int tile_d0, unsigned int tile_d1,
                                            unsigned int row_stride_elems)
{
    union { unsigned long long q[2]; u32x4 v; } g0;
    g0.q[0] = 1ull                                            // count=1, no gather
            | ((unsigned long long)lds_off << 32);            // lds_addr
    g0.q[1] = ((unsigned long long)(uintptr_t)gaddr & 0x01FFFFFFFFFFFFFFull)
            | (2ull << 62);                                   // type=2 ("image")

    const unsigned long long tdim0  = row_stride_elems;       // no OOB clamp needed
    const unsigned long long tdim1  = 1ull << 30;
    const unsigned long long str0   = row_stride_elems;       // dim0 stride (elems)
    union { unsigned long long q[4]; i32x8 v; } g1;
    g1.q[0] = (1ull << 16)                                    // data_size = 1 -> 2 bytes
            | (tdim0 << 48);                                  // tensor_dim0[15:0]
    g1.q[1] = (tdim0 >> 16)                                   // tensor_dim0[31:16]
            | (tdim1 << 16)                                   // tensor_dim1
            | ((unsigned long long)tile_d0 << 48);            // tile_dim0
    g1.q[2] = (unsigned long long)tile_d1                     // tile_dim1
            | (str0 << 32);                                   // tensor_dim0_stride[31:0]
    g1.q[3] = (str0 >> 32);                                   // stride[47:32]; dim1 stride unused (tile_dim2=0)

    const i32x4 z4 = (i32x4){0, 0, 0, 0};
#if __clang_major__ >= 23
    const i32x8 z8 = (i32x8){0, 0, 0, 0, 0, 0, 0, 0};
    __builtin_amdgcn_tensor_load_to_lds(g0.v, g1.v, z4, z4, z8, 0);
#else
    __builtin_amdgcn_tensor_load_to_lds(g0.v, g1.v, z4, z4, 0);
#endif
}
#endif // TDM_OK

// ---------------------------------------------------------------------------
// Generic bf16 WMMA GEMM: C[M,N] = act(A[M,K] @ W[N,K]^T + bias)
// A row-major MxK bf16, W row-major NxK bf16.
// Block: 256 threads = 8 waves (2 waveM x 4 waveN); block tile 128M x 256N;
// wave tile 64x64 = 4x4 WMMA 16x16x32 bf16 tiles.
// TDM path: wave 0 DMAs double-buffered A(128x32)/B(256x32) tiles into LDS,
// all waves read fragments from LDS.  Fallback: direct global fragment loads.
// M % 128 == 0, N % 256 == 0, K % 32 == 0 (true for all uses here).
// ---------------------------------------------------------------------------
__global__ __launch_bounds__(256) void gemm_bf16_kernel(
    const __hip_bfloat16* __restrict__ A,
    const __hip_bfloat16* __restrict__ W,
    const float* __restrict__ bias,
    float* __restrict__ Cf,
    __hip_bfloat16* __restrict__ Cb,
    int M, int N, int K, int do_gelu)
{
    const int lane   = threadIdx.x & 31;
    const int wave   = threadIdx.x >> 5;
    const int waveM  = wave >> 2;            // 0..1
    const int waveN  = wave & 3;             // 0..3
    const int bm_blk = blockIdx.y * BM;
    const int bn_blk = blockIdx.x * BN;
    const int lr     = lane & 15;            // row within a 16-wide tile
    const int khalf  = (lane >> 4) << 3;     // lanes 0-15 -> K+0, lanes 16-31 -> K+8

    v8f accs[4][4];
#pragma unroll
    for (int m = 0; m < 4; ++m)
#pragma unroll
        for (int n = 0; n < 4; ++n) accs[m][n] = (v8f){};

#if TDM_OK
    __shared__ __align__(128) __hip_bfloat16 ldsA[2][BM * BK];   // 8 KB x2
    __shared__ __align__(128) __hip_bfloat16 ldsB[2][BN * BK];   // 16 KB x2
    const bool issuer = (wave == 0);
    if (issuer) {
        tdm_load_2d(A + (size_t)bm_blk * K, (unsigned int)(uintptr_t)&ldsA[0][0], BK, BM, K);
        tdm_load_2d(W + (size_t)bn_blk * K, (unsigned int)(uintptr_t)&ldsB[0][0], BK, BN, K);
    }
    for (int kk = 0; kk < K; kk += BK) {
        const int buf = (kk >> 5) & 1;
        if (issuer) {
            if (kk + BK < K) {
                tdm_load_2d(A + (size_t)bm_blk * K + kk + BK,
                            (unsigned int)(uintptr_t)&ldsA[buf ^ 1][0], BK, BM, K);
                tdm_load_2d(W + (size_t)bn_blk * K + kk + BK,
                            (unsigned int)(uintptr_t)&ldsB[buf ^ 1][0], BK, BN, K);
                __builtin_amdgcn_s_wait_tensorcnt(2);  // current buf's pair done
            } else {
                __builtin_amdgcn_s_wait_tensorcnt(0);
            }
        }
        __syncthreads();                               // buf data visible to all waves

        FragBF a[4], b[4];
#pragma unroll
        for (int m = 0; m < 4; ++m) {
            const __hip_bfloat16* p = &ldsA[buf][(waveM * 64 + m * 16 + lr) * BK + khalf];
            a[m].q[0] = *(const uint4*)p;
            a[m].q[1] = *(const uint4*)(p + 16);
        }
#pragma unroll
        for (int n = 0; n < 4; ++n) {
            const __hip_bfloat16* p = &ldsB[buf][(waveN * 64 + n * 16 + lr) * BK + khalf];
            b[n].q[0] = *(const uint4*)p;
            b[n].q[1] = *(const uint4*)(p + 16);
        }
#pragma unroll
        for (int m = 0; m < 4; ++m)
#pragma unroll
            for (int n = 0; n < 4; ++n)
                accs[m][n] = __builtin_amdgcn_wmma_f32_16x16x32_bf16(
                    false, a[m].v, false, b[n].v, (short)0, accs[m][n], false, false);

        __syncthreads();                               // readers done before buf reuse
    }
#else
    // Fallback: direct global fragment loads (16-bit A/B VGPR striping:
    // lanes 0-15 take K[0:8)+K[16:24), lanes 16-31 take K[8:16)+K[24:32)).
    for (int kk = 0; kk < K; kk += BK) {
        FragBF a[4], b[4];
#pragma unroll
        for (int m = 0; m < 4; ++m) {
            const __hip_bfloat16* p = A + (size_t)(bm_blk + waveM * 64 + m * 16 + lr) * K + kk + khalf;
            a[m].q[0] = *(const uint4*)p;
            a[m].q[1] = *(const uint4*)(p + 16);
        }
#pragma unroll
        for (int n = 0; n < 4; ++n) {
            const __hip_bfloat16* p = W + (size_t)(bn_blk + waveN * 64 + n * 16 + lr) * K + kk + khalf;
            b[n].q[0] = *(const uint4*)p;
            b[n].q[1] = *(const uint4*)(p + 16);
        }
#pragma unroll
        for (int m = 0; m < 4; ++m)
#pragma unroll
            for (int n = 0; n < 4; ++n)
                accs[m][n] = __builtin_amdgcn_wmma_f32_16x16x32_bf16(
                    false, a[m].v, false, b[n].v, (short)0, accs[m][n], false, false);
    }
#endif

    // C/D layout: VGPR r -> row m = r (lanes 0-15) or 8+r (lanes 16-31), col = lane%16
    const int rbase = (lane >> 4) << 3;
#pragma unroll
    for (int m = 0; m < 4; ++m) {
#pragma unroll
        for (int n = 0; n < 4; ++n) {
            const int row0 = bm_blk + waveM * 64 + m * 16 + rbase;
            const int col  = bn_blk + waveN * 64 + n * 16 + lr;
            const float bc = bias ? bias[col] : 0.0f;
#pragma unroll
            for (int r = 0; r < 8; ++r) {
                float v = accs[m][n][r] + bc;
                if (do_gelu) v = gelu_f(v);
                const size_t idx = (size_t)(row0 + r) * (size_t)N + (size_t)col;
                if (Cf) Cf[idx] = v;
                if (Cb) Cb[idx] = __float2bfloat16(v);
            }
        }
    }
}

// ---------------------------------------------------------------------------
// Precompute kernels
// ---------------------------------------------------------------------------
__global__ void gather_emb_kernel(const int* __restrict__ x_seq,
                                  const float* __restrict__ emb,
                                  __hip_bfloat16* __restrict__ out)
{
    const size_t i = (size_t)blockIdx.x * blockDim.x + threadIdx.x; // T*HID threads
    const int t = (int)(i >> 10);
    const int c = (int)(i & (HID - 1));
    out[i] = __float2bfloat16(emb[(size_t)x_seq[t] * HID + c]);
}

__global__ void cvt_bf16_kernel(const float* __restrict__ in,
                                __hip_bfloat16* __restrict__ out)
{
    const size_t i = (size_t)blockIdx.x * blockDim.x + threadIdx.x;
    out[i] = __float2bfloat16(in[i]);
}

// At[l][k][r] = A[l][r][k]  (f32, for coalesced scan reads)
__global__ void transpose_a_kernel(const float* __restrict__ in,
                                   float* __restrict__ out)
{
    const size_t i = (size_t)blockIdx.x * blockDim.x + threadIdx.x; // LAYERS*HID*HID
    const int l = (int)(i >> 20);
    const int k = (int)((i >> 10) & (HID - 1));
    const int r = (int)(i & (HID - 1));
    out[i] = in[(size_t)l * HID * HID + (size_t)r * HID + k];
}

// B0t[n][k] = b_mats[0][k][n] as bf16 (W operand of the U0 GEMM)
__global__ void transpose_b0_kernel(const float* __restrict__ in,
                                    __hip_bfloat16* __restrict__ out)
{
    const size_t i = (size_t)blockIdx.x * blockDim.x + threadIdx.x; // HID*HID
    const int n = (int)(i >> 10);
    const int k = (int)(i & (HID - 1));
    out[i] = __float2bfloat16(in[(size_t)k * HID + n]);
}

// Zero the double-buffered h state and the grid-barrier counter each call.
__global__ void init_scan_kernel(float* __restrict__ hb, unsigned int* __restrict__ bar)
{
    const int i = blockIdx.x * blockDim.x + threadIdx.x; // 2*LAYERS*HID = 8192 threads
    hb[i] = 0.0f;
    if (i == 0) *bar = 0u;
}

// ---------------------------------------------------------------------------
// Persistent sequential scan.  8192 threads: 8 k-slices per hidden row.
// Per step t, per layer l:  h = A_l @ h_prev + u ;  x = gelu(h) + x_in.
// Layer 0's u is the precomputed U0 = X1 @ B0 (WMMA GEMM); layers 1-3 compute
// u = B_l^T x on the fly.  Grid-wide barrier = monotonic atomic counter.
// ---------------------------------------------------------------------------
__global__ __launch_bounds__(SCAN_THREADS) void scan_kernel(
    const float* __restrict__ X1,     // TSTEPS x HID (gelu'd proj_in output)
    const float* __restrict__ U0,     // TSTEPS x HID
    const float* __restrict__ At,     // LAYERS x HID x HID  (A transposed: [l][k][r])
    const float* __restrict__ Bm,     // LAYERS x HID x HID  (original b_mats: [l][k][n])
    float* __restrict__ hb,           // 2 x LAYERS x HID (double buffer by t parity)
    float* __restrict__ xping,        // 2 x HID (ping-pong by layer parity)
    __hip_bfloat16* __restrict__ Xout,// TSTEPS x HID
    unsigned int* __restrict__ bar)
{
    const int gtid = blockIdx.x * SCAN_THREADS + threadIdx.x;
    const int r  = gtid >> 3;        // hidden row 0..1023
    const int ks = gtid & 7;         // k-slice
    const int k0 = ks * 128;
    unsigned int bar_it = 0;

    for (int t = 0; t < TSTEPS; ++t) {
        const int wp = t & 1, rp = wp ^ 1;
        if (t + 1 < TSTEPS) {        // global_prefetch_b8 of next step's vectors
            __builtin_prefetch(X1 + (size_t)(t + 1) * HID + (ks << 7), 0, 1);
            __builtin_prefetch(U0 + (size_t)(t + 1) * HID + (ks << 7), 0, 1);
        }
        for (int l = 0; l < LAYERS; ++l) {
            const float* xin   = (l == 0) ? (X1 + (size_t)t * HID)
                                          : (xping + (size_t)((l - 1) & 1) * HID);
            const float* hprev = hb + (size_t)(rp * LAYERS + l) * HID;
            const float* Atl   = At + (size_t)l * HID * HID;
            const float* Bl    = Bm + (size_t)l * HID * HID;

            float acc = 0.0f;
            if (l == 0) {
                for (int k = k0; k < k0 + 128; ++k)
                    acc = fmaf(Atl[(size_t)k * HID + r], hprev[k], acc);
            } else {
                for (int k = k0; k < k0 + 128; ++k) {
                    acc = fmaf(Atl[(size_t)k * HID + r], hprev[k], acc);
                    acc = fmaf(Bl[(size_t)k * HID + r], xin[k], acc);
                }
            }
            // reduce the 8 k-slices (lanes 8i..8i+7 of the wave)
            acc += __shfl_xor(acc, 4, 32);
            acc += __shfl_xor(acc, 2, 32);
            acc += __shfl_xor(acc, 1, 32);

            if (ks == 0) {
                const float h = acc + ((l == 0) ? U0[(size_t)t * HID + r] : 0.0f);
                hb[(size_t)(wp * LAYERS + l) * HID + r] = h;
                const float xnew = gelu_f(h) + xin[r];
                if (l < LAYERS - 1) xping[(size_t)(l & 1) * HID + r] = xnew;
                else                Xout[(size_t)t * HID + r] = __float2bfloat16(xnew);
            }

            // ---- grid barrier (monotonic counter, no reset) ----
            __threadfence();
            __syncthreads();
            ++bar_it;
            if (threadIdx.x == 0) {
                atomicAdd(bar, 1u);
                const unsigned int target = bar_it * (unsigned int)SCAN_BLOCKS;
                while (__hip_atomic_load(bar, __ATOMIC_RELAXED, __HIP_MEMORY_SCOPE_AGENT) < target)
                    __builtin_amdgcn_s_sleep(2);
            }
            __syncthreads();
            __threadfence();
        }
    }
}

// ---------------------------------------------------------------------------
extern "C" void kernel_launch(void* const* d_in, const int* in_sizes, int n_in,
                              void* d_out, int out_size, void* d_ws, size_t ws_size,
                              hipStream_t stream)
{
    const int*   x_seq  = (const int*)  d_in[0];
    const float* emb    = (const float*)d_in[1];
    const float* w_in   = (const float*)d_in[2];
    const float* b_in   = (const float*)d_in[3];
    const float* w_out  = (const float*)d_in[4];
    const float* b_out  = (const float*)d_in[5];
    const float* a_mats = (const float*)d_in[6];
    const float* b_mats = (const float*)d_in[7];
    float* out = (float*)d_out;

    char* ws = (char*)d_ws;
    size_t off = 0;
    auto walloc = [&](size_t bytes) -> void* {
        void* p = ws + off;
        off = (off + bytes + 255) & ~(size_t)255;
        return p;
    };

    __hip_bfloat16* Xg     = (__hip_bfloat16*)walloc((size_t)TSTEPS * HID * 2);          // gathered emb (bf16)
    __hip_bfloat16* Wi_bf  = (__hip_bfloat16*)walloc((size_t)HID * HID * 2);             // w_in bf16
    float*          X1f    = (float*)         walloc((size_t)TSTEPS * HID * 4);          // gelu(proj_in) f32
    __hip_bfloat16* X1b    = (__hip_bfloat16*)walloc((size_t)TSTEPS * HID * 2);          // same, bf16
    __hip_bfloat16* B0t    = (__hip_bfloat16*)walloc((size_t)HID * HID * 2);             // b_mats[0]^T bf16
    float*          U0     = (float*)         walloc((size_t)TSTEPS * HID * 4);          // X1 @ B0
    float*          At     = (float*)         walloc((size_t)LAYERS * HID * HID * 4);    // a_mats transposed
    __hip_bfloat16* Wout_b = (__hip_bfloat16*)walloc((size_t)OUTDIM * HID * 2);          // w_out bf16
    __hip_bfloat16* Xout_b = (__hip_bfloat16*)walloc((size_t)TSTEPS * HID * 2);          // scan output bf16
    float*          hb     = (float*)         walloc((size_t)2 * LAYERS * HID * 4);      // h double buffer
    float*          xping  = (float*)         walloc((size_t)2 * HID * 4);               // x ping-pong
    unsigned int*   bar    = (unsigned int*)  walloc(256);                               // grid barrier

    // --- precompute ---
    gather_emb_kernel<<<(TSTEPS * HID) / 256, 256, 0, stream>>>(x_seq, emb, Xg);
    cvt_bf16_kernel<<<(HID * HID) / 256, 256, 0, stream>>>(w_in, Wi_bf);
    cvt_bf16_kernel<<<(OUTDIM * HID) / 256, 256, 0, stream>>>(w_out, Wout_b);
    transpose_a_kernel<<<(LAYERS * HID * HID) / 256, 256, 0, stream>>>(a_mats, At);
    transpose_b0_kernel<<<(HID * HID) / 256, 256, 0, stream>>>(b_mats, B0t);
    init_scan_kernel<<<(2 * LAYERS * HID) / 256, 256, 0, stream>>>(hb, bar);

    // --- GEMM 1: X1 = gelu(Xg @ w_in^T + b_in) (f32 + bf16 outputs) ---
    {
        dim3 grid(HID / BN, TSTEPS / BM);
        gemm_bf16_kernel<<<grid, 256, 0, stream>>>(Xg, Wi_bf, b_in, X1f, X1b,
                                                   TSTEPS, HID, HID, /*gelu=*/1);
    }
    // --- GEMM U0: U0 = X1 @ B0  (= X1 @ (B0^T)^T) ---
    {
        dim3 grid(HID / BN, TSTEPS / BM);
        gemm_bf16_kernel<<<grid, 256, 0, stream>>>(X1b, B0t, nullptr, U0, nullptr,
                                                   TSTEPS, HID, HID, /*gelu=*/0);
    }
    // --- sequential scan (persistent, grid-synced) ---
    scan_kernel<<<SCAN_BLOCKS, SCAN_THREADS, 0, stream>>>(X1f, U0, At, b_mats,
                                                          hb, xping, Xout_b, bar);
    // --- GEMM 2: out = Xout @ w_out^T + b_out ---
    {
        dim3 grid(OUTDIM / BN, TSTEPS / BM);
        gemm_bf16_kernel<<<grid, 256, 0, stream>>>(Xout_b, Wout_b, b_out, out, nullptr,
                                                   TSTEPS, OUTDIM, HID, /*gelu=*/0);
    }
    (void)in_sizes; (void)n_in; (void)out_size; (void)ws_size;
}
